// QRNNModel_47588237640330
// MI455X (gfx1250) — matrix operand order repaired and checked by reading, not compiled, not run on hardware
//
#include <hip/hip_runtime.h>
#include <cstdint>

// ---------------------------------------------------------------------------
// QRNN on MI455X (gfx1250, wave32).
//  - x, W, V pre-packed to bf16 once (RNE); W/V stored K-pair-major so LDS
//    tiles are raw copies in WMMA fragment layout.
//  - GEMM staging is pure global_load_async_to_lds_b128 (ASYNCcnt), double-
//    buffered: async copy of step k+1 overlaps WMMA of step k.
//  - Dominant GEMM via v_wmma_f32_16x16x32_bf16 (fp32 accumulate).
//  - fo_pool collapsed to last-state fold via segmented scan.
// ---------------------------------------------------------------------------

typedef __attribute__((ext_vector_type(16))) __bf16 v16bf;
typedef __attribute__((ext_vector_type(8)))  float  v8f;

union Frag16 { unsigned int u[8]; v16bf v; };

#define S_LEN   2048
#define DIN     512
#define NTOT    3072   // W,V column count
#define NFZ     2048   // f+z columns
#define CHB     4      // batches per chunk
#define CHTOK   (CHB * S_LEN)   // 8192 tokens per chunk

#define BM      256    // block tile M (tokens)
#define BN      128    // block tile N (channels)
#define AROWU   20     // A tile row stride in u32 (80 B, 16B-aligned)
#define BROWU   136    // B tile pair-row stride in u32 (544 B, 16B-aligned)
#define ABUFB   (BM * AROWU * 4)    // 20480 B per A buffer
#define BBUFB   (16 * BROWU * 4)    // 8704 B per B buffer

__device__ __forceinline__ unsigned int pack_bf16x2(float a, float b) {
    unsigned int ua = __float_as_uint(a), ub = __float_as_uint(b);
    unsigned int ra = (ua + 0x7FFFu + ((ua >> 16) & 1u)) >> 16;   // RNE
    unsigned int rb = (ub + 0x7FFFu + ((ub >> 16) & 1u)) >> 16;
    return (ra & 0xFFFFu) | (rb << 16);
}

// Cheap activations on hardware transcendentals (v_exp_f32 + v_rcp_f32).
__device__ __forceinline__ float fast_sigmoid(float x) {
    return __builtin_amdgcn_rcpf(1.f + __expf(-x));
}
__device__ __forceinline__ float fast_tanh(float x) {
    return 1.f - 2.f * __builtin_amdgcn_rcpf(__expf(2.f * x) + 1.f);
}

// CDNA5 async global->LDS copy, tracked by ASYNCcnt (ISA 15.18.3 op 98).
__device__ __forceinline__ void async_ld_b128(unsigned lds_off, const void* gaddr) {
    asm volatile("global_load_async_to_lds_b128 %0, %1, off"
                 :: "v"(lds_off), "v"(gaddr)
                 : "memory");
}
__device__ __forceinline__ void wait_asynccnt0() {
    asm volatile("s_wait_asynccnt 0" ::: "memory");
}

// ---------------------------------------------------------------------------
// Pack W/V (f,z columns) to bf16 K-pair-major: dst[p][n] = {M[2p][n], M[2p+1][n]}
// dst is [256][2048] u32. One-time pass.
// ---------------------------------------------------------------------------
__global__ __launch_bounds__(256)
void qrnn_pack_wv(const float* __restrict__ src, unsigned int* __restrict__ dst)
{
    int id = blockIdx.x * 256 + threadIdx.x;   // 256*2048 words
    int p = id >> 11, n = id & 2047;
    dst[id] = pack_bf16x2(src[(size_t)(2 * p) * NTOT + n],
                          src[(size_t)(2 * p + 1) * NTOT + n]);
}

// Pack one x chunk to bf16 (row-major, pairs packed in u32). Per-chunk pass.
__global__ __launch_bounds__(256)
void qrnn_pack_x(const float* __restrict__ xc, unsigned int* __restrict__ xbfw)
{
    int id = blockIdx.x * 256 + threadIdx.x;   // CHTOK*DIN/2 words
    const float2 v = *(const float2*)(xc + (size_t)id * 2);
    xbfw[id] = pack_bf16x2(v.x, v.y);
}

// ---------------------------------------------------------------------------
// Kernel 1: block tile 256(M) x 128(N), K = 1024 (phase 0: x@W, phase 1: xprev@V)
// 8 waves = 4(M) x 2(N); wave tile 64x64 = 4x4 WMMA accumulators.
// Both tiles staged bf16 via async b128, double-buffered software pipeline.
// Epilogue: +Vb, sigmoid(f) / tanh(z), store to act[8192][2048].
// ---------------------------------------------------------------------------
__global__ __launch_bounds__(256)
void qrnn_gemm_fz(const unsigned short* __restrict__ xbf,  // [8192][512] bf16
                  const unsigned int* __restrict__ Wpk,    // [256][2048] u32 pairs
                  const unsigned int* __restrict__ Vpk,    // [256][2048] u32 pairs
                  const float* __restrict__ Vb,            // [3072]
                  float* __restrict__ act)                 // [8192][2048] f32
{
    __shared__ __align__(16) unsigned int sA[2][BM * AROWU];  // bf16 A tiles
    __shared__ __align__(16) unsigned int sB[2][16 * BROWU];  // bf16-pair B tiles

    const int t    = threadIdx.x;
    const int lane = t & 31;
    const int wave = t >> 5;
    const int wm   = wave & 3;        // M quarter (64 rows)
    const int wn   = wave >> 2;       // N half (64 cols)
    const int hs   = lane >> 4;       // half-select (lanes 16..31)
    const int mRow = lane & 15;

    const int g0    = blockIdx.x * BM;   // chunk-local token base
    const int nBase = blockIdx.y * BN;
    const unsigned sA_lds = (unsigned)(uintptr_t)&sA[0][0];
    const unsigned sB_lds = (unsigned)(uintptr_t)&sB[0][0];

    // per-thread staging coordinates (loop-invariant)
    const unsigned aOff = (unsigned)((t >> 2) * 80 + (t & 3) * 16);   // bytes
    const unsigned bOff = (unsigned)((t >> 5) * 544 + (t & 31) * 16); // bytes

    // Issue all async copies for K-step `ks` into buffer (ks&1).
    auto stage = [&](int ks) {
        const int phase = ks >> 4;                 // 0: x@W, 1: xprev@V (uniform)
        const int buf   = ks & 1;
        const bool bnd  = (phase == 1) && ((g0 & (S_LEN - 1)) == 0);
        // A: 1024 b128 chunks (4/thread); row m = (t>>2) + 64i, 8 bf16 per chunk
        const unsigned short* asrc = xbf + (size_t)(g0 + (t >> 2)) * DIN
                                         + (ks & 15) * 32 + (t & 3) * 8
                                         - (phase ? 512 : 0);     // token m-1
        const unsigned aDst = sA_lds + (unsigned)buf * ABUFB + aOff;
        #pragma unroll
        for (int i = 0; i < 4; ++i) {
            if (i == 0 && bnd && t < 4) {
                // token with s==0: xprev row is zero (also guards x underrun)
                *(uint4*)&sA[buf][t * 4] = uint4{0u, 0u, 0u, 0u};  // row 0
            } else {
                async_ld_b128(aDst + (unsigned)(i * 64 * 80),
                              asrc + (size_t)i * 64 * DIN);
            }
        }
        // B: 512 b128 chunks (2/thread); pair-row p = (t>>5) + 8i, 4 u32/chunk
        const unsigned int* bsrc = (phase ? Vpk : Wpk)
            + (size_t)((ks & 15) * 16 + (t >> 5)) * 2048 + nBase + (t & 31) * 4;
        const unsigned bDst = sB_lds + (unsigned)buf * BBUFB + bOff;
        #pragma unroll
        for (int i = 0; i < 2; ++i)
            async_ld_b128(bDst + (unsigned)(i * 8 * 544),
                          bsrc + (size_t)i * 8 * 2048);
    };

    v8f acc[4][4] = {};

    stage(0);
    wait_asynccnt0();
    __syncthreads();

    for (int ks = 0; ks < 32; ++ks) {          // 32 K-steps of 32
        if (ks + 1 < 32) stage(ks + 1);        // async copy overlaps compute

        // ---- fragments (contiguous k = 16*hs + s in A and B) + 16 WMMAs ----
        const unsigned int* sAc = &sA[ks & 1][0];
        const unsigned int* sBc = &sB[ks & 1][0];
        Frag16 a[4], b[4];
        #pragma unroll
        for (int j = 0; j < 8; ++j) {
            int pa = hs * 8 + j;                       // A K-pair index
            #pragma unroll
            for (int i = 0; i < 4; ++i)
                a[i].u[j] = sAc[(wm * 64 + i * 16 + mRow) * AROWU + pa];
            int pb = hs * 8 + j;                       // B: lanes16-31 -> K=16..31
            #pragma unroll
            for (int jj = 0; jj < 4; ++jj)
                b[jj].u[j] = sBc[pb * BROWU + wn * 64 + jj * 16 + mRow];
        }
        #pragma unroll
        for (int i = 0; i < 4; ++i)
            #pragma unroll
            for (int jj = 0; jj < 4; ++jj)
                acc[i][jj] = __builtin_amdgcn_wmma_f32_16x16x32_bf16(
                    false, a[i].v, false, b[jj].v, (short)0, acc[i][jj],
                    false, false);

        wait_asynccnt0();                      // next buffer landed
        __syncthreads();                       // everyone done with current
    }

    // ---- epilogue: bias + activation, write f / tanh(z) ----
    // N-tile is 128-wide and the f/z boundary (1024) is a multiple of 128,
    // so the activation choice is block-uniform.
    const bool isF = (nBase < 1024);
    #pragma unroll
    for (int i = 0; i < 4; ++i) {
        int mBase = g0 + wm * 64 + i * 16;
        #pragma unroll
        for (int jj = 0; jj < 4; ++jj) {
            int col  = nBase + wn * 64 + jj * 16 + mRow;  // C/D: N = lane&15
            float bias = Vb[col];
            #pragma unroll
            for (int v = 0; v < 8; ++v) {
                int row = mBase + v + hs * 8;             // C/D: M = vgpr + 8*(lane>=16)
                float val = acc[i][jj][v] + bias;
                float r = isF ? fast_sigmoid(val)         // f gate
                              : fast_tanh(val);           // tanh(z)
                act[(size_t)row * NFZ + col] = r;
            }
        }
    }
}

// ---------------------------------------------------------------------------
// Kernel 2a: segmented scan. Per (b,k): 8 segments of 256 steps.
// Monoid element (F = prod f, C = fold of c<-f*c+(1-f)*tz).
// ---------------------------------------------------------------------------
__global__ __launch_bounds__(256)
void qrnn_scan_seg(const float* __restrict__ act,
                   float* __restrict__ Fseg, float* __restrict__ Cseg)
{
    int id  = blockIdx.x * 256 + threadIdx.x;   // 32768 threads
    int k   = id & 1023;
    int seg = (id >> 10) & 7;
    int b   = id >> 13;
    const float* base = act + (size_t)(b * S_LEN + seg * 256) * NFZ;
    float F = 1.f, C = 0.f;
    for (int s = 0; s < 256; ++s) {
        float f  = base[(size_t)s * NFZ + k];
        float tz = base[(size_t)s * NFZ + 1024 + k];
        C = f * C + (1.f - f) * tz;
        F *= f;
    }
    int o = (b * 8 + seg) * 1024 + k;
    Fseg[o] = F; Cseg[o] = C;
}

// Kernel 2b: serial combine of the 8 segments -> c at s = S-1.
__global__ __launch_bounds__(256)
void qrnn_scan_combine(const float* __restrict__ Fseg, const float* __restrict__ Cseg,
                       float* __restrict__ c_last, int bgbase)
{
    int id = blockIdx.x * 256 + threadIdx.x;    // 4096 threads
    int k = id & 1023, b = id >> 10;
    float c = 0.f;
    for (int seg = 0; seg < 8; ++seg) {
        int o = (b * 8 + seg) * 1024 + k;
        c = Fseg[o] * c + Cseg[o];
    }
    c_last[(size_t)(bgbase + b) * 1024 + k] = c;
}

// ---------------------------------------------------------------------------
// Kernel 3a: o-gate at last timestep only + h = c * sigmoid(o_pre).
// 32 rows -> negligible FLOPs, plain dot products.
// ---------------------------------------------------------------------------
__global__ __launch_bounds__(256)
void qrnn_head_oh(const float* __restrict__ x, const float* __restrict__ W,
                  const float* __restrict__ V, const float* __restrict__ Vb,
                  const float* __restrict__ c_last, float* __restrict__ h)
{
    int b = blockIdx.x;
    const float* xl = x + ((size_t)b * S_LEN + (S_LEN - 1)) * DIN;
    const float* xp = x + ((size_t)b * S_LEN + (S_LEN - 2)) * DIN;
    for (int k = threadIdx.x; k < 1024; k += 256) {
        int col = 2048 + k;
        float acc = Vb[col];
        for (int d = 0; d < DIN; ++d)
            acc += xl[d] * W[(size_t)d * NTOT + col] + xp[d] * V[(size_t)d * NTOT + col];
        h[(size_t)b * 1024 + k] = c_last[(size_t)b * 1024 + k] * fast_sigmoid(acc);
    }
}

// Kernel 3b: fully-connected layer, optional relu. 32 rows -> trivial cost.
__global__ __launch_bounds__(256)
void qrnn_fc(const float* __restrict__ in, const float* __restrict__ Wm,
             const float* __restrict__ bm, float* __restrict__ out,
             int Kdim, int Ndim, int relu)
{
    int b = blockIdx.y;
    int j = blockIdx.x * 256 + threadIdx.x;
    if (j >= Ndim) return;
    float acc = bm[j];
    const float* row = in + (size_t)b * Kdim;
    for (int i = 0; i < Kdim; ++i)
        acc += row[i] * Wm[(size_t)i * Ndim + j];
    if (relu) acc = fmaxf(acc, 0.f);
    out[(size_t)b * Ndim + j] = acc;
}

// ---------------------------------------------------------------------------
extern "C" void kernel_launch(void* const* d_in, const int* in_sizes, int n_in,
                              void* d_out, int out_size, void* d_ws, size_t ws_size,
                              hipStream_t stream)
{
    const float* x  = (const float*)d_in[0];
    const float* W  = (const float*)d_in[1];
    const float* V  = (const float*)d_in[2];
    const float* Vb = (const float*)d_in[3];
    const float* W0 = (const float*)d_in[4];
    const float* b0 = (const float*)d_in[5];
    const float* W1 = (const float*)d_in[6];
    const float* b1 = (const float*)d_in[7];
    const float* W2 = (const float*)d_in[8];
    const float* b2 = (const float*)d_in[9];
    float* out = (float*)d_out;

    char* ws = (char*)d_ws;
    size_t off = 0;
    float*        act    = (float*)(ws + off);        off += (size_t)CHTOK * NFZ * 4;   // 64 MB
    unsigned int* xbf    = (unsigned int*)(ws + off); off += (size_t)CHTOK * DIN * 2;   // 8 MB
    unsigned int* Wpk    = (unsigned int*)(ws + off); off += (size_t)256 * 2048 * 4;    // 2 MB
    unsigned int* Vpk    = (unsigned int*)(ws + off); off += (size_t)256 * 2048 * 4;    // 2 MB
    float*        Fseg   = (float*)(ws + off);        off += (size_t)CHB * 8 * 1024 * 4;
    float*        Cseg   = (float*)(ws + off);        off += (size_t)CHB * 8 * 1024 * 4;
    float*        c_last = (float*)(ws + off);        off += (size_t)32 * 1024 * 4;
    float*        h      = (float*)(ws + off);        off += (size_t)32 * 1024 * 4;
    float*        q0     = (float*)(ws + off);        off += (size_t)32 * 1024 * 4;
    float*        q1     = (float*)(ws + off);        off += (size_t)32 * 1024 * 4;
    (void)ws_size; (void)in_sizes; (void)n_in; (void)out_size;

    // One-time weight pack (f,z columns only).
    qrnn_pack_wv<<<2048, 256, 0, stream>>>(W, Wpk);
    qrnn_pack_wv<<<2048, 256, 0, stream>>>(V, Vpk);

    // 4-batch chunks keep workspace at ~77 MB; batches are independent
    // through the scan, so chunking is exact.
    for (int chunk = 0; chunk < 8; ++chunk) {
        const float* xc = x + (size_t)chunk * CHB * S_LEN * DIN;
        qrnn_pack_x<<<CHTOK * DIN / 2 / 256, 256, 0, stream>>>(xc, xbf);
        qrnn_gemm_fz<<<dim3(CHTOK / BM, NFZ / BN), 256, 0, stream>>>(
            (const unsigned short*)xbf, Wpk, Vpk, Vb, act);
        qrnn_scan_seg<<<128, 256, 0, stream>>>(act, Fseg, Cseg);
        qrnn_scan_combine<<<16, 256, 0, stream>>>(Fseg, Cseg, c_last, chunk * CHB);
    }
    qrnn_head_oh<<<32, 256, 0, stream>>>(x, W, V, Vb, c_last, h);
    qrnn_fc<<<dim3(4, 32), 256, 0, stream>>>(h,  W0, b0, q0, 1024, 1024, 1);
    qrnn_fc<<<dim3(4, 32), 256, 0, stream>>>(q0, W1, b1, q1, 1024, 1024, 1);
    qrnn_fc<<<dim3(1, 32), 256, 0, stream>>>(q1, W2, b2, out, 1024, 128, 0);
}
